// GatedGNN_25074019074619
// MI455X (gfx1250) — matrix operand order repaired and checked by reading, not compiled, not run on hardware
//
#include <hip/hip_runtime.h>
#include <hip/hip_bf16.h>
#include <math.h>

// ---------------------------------------------------------------------------
// Fixed structural constants of the reference model (C, T are architectural).
// N, E, B, NPG are derived from in_sizes/out_size at launch.
// ---------------------------------------------------------------------------
constexpr int C = 128;    // channels
constexpr int T = 2048;   // embedding table rows / logits width
constexpr int WPB = 8;    // waves per block (256 threads, wave32)

typedef __attribute__((ext_vector_type(16))) __bf16 v16bf;
typedef __attribute__((ext_vector_type(8)))  __bf16 v8bf;
typedef __attribute__((ext_vector_type(4)))  __bf16 v4bf;
typedef __attribute__((ext_vector_type(8)))  float  v8f;

__device__ __forceinline__ __bf16 f2bf(float f) {
  // round-to-nearest-even fp32 -> bf16
  unsigned u = __builtin_bit_cast(unsigned, f);
  unsigned r = u + 0x7FFFu + ((u >> 16) & 1u);
  unsigned short hs = (unsigned short)(r >> 16);
  return __builtin_bit_cast(__bf16, hs);
}
__device__ __forceinline__ float bf2f(__bf16 b) {
  unsigned u = (unsigned)__builtin_bit_cast(unsigned short, b) << 16;
  return __builtin_bit_cast(float, u);
}
__device__ __forceinline__ v8f zero8() {
  v8f z = {0.f, 0.f, 0.f, 0.f, 0.f, 0.f, 0.f, 0.f};
  return z;
}

// A-fragment (16x32 bf16) from a row-major bf16 matrix (works for LDS too).
// ISA layout: lane L, element e -> row = r0+(L&15),
//             k = k0 + (e>>3)*16 + (L>>4)*8 + (e&7)  => two contiguous 16B runs.
__device__ __forceinline__ v16bf load_a(const __bf16* M, int ld,
                                        int r0, int k0, int lane) {
  const __bf16* p = M + (size_t)(r0 + (lane & 15)) * ld + k0 + ((lane >> 4) << 3);
  v8bf lo = *(const v8bf*)p;
  v8bf hi = *(const v8bf*)(p + 16);
  return __builtin_shufflevector(lo, hi, 0, 1, 2, 3, 4, 5, 6, 7,
                                 8, 9, 10, 11, 12, 13, 14, 15);
}

// B-fragment (32x16 bf16) for X @ W^T, W row-major bf16 (out x in):
// lane L, element e -> n = n0+(L&15), k = k0 + (e>>3)*16 + (L>>4)*8 + (e&7)
// => two contiguous 16B runs of a weight row (same k order as load_a).
__device__ __forceinline__ v16bf load_b(const __bf16* W, int ld,
                                        int n0, int k0, int lane) {
  const __bf16* p = W + (size_t)(n0 + (lane & 15)) * ld + k0 + ((lane >> 4) << 3);
  v8bf lo = *(const v8bf*)p;
  v8bf hi = *(const v8bf*)(p + 16);
  return __builtin_shufflevector(lo, hi, 0, 1, 2, 3, 4, 5, 6, 7,
                                 8, 9, 10, 11, 12, 13, 14, 15);
}

__device__ __forceinline__ v8f wmma_bf16(v16bf a, v16bf b, v8f c) {
  return __builtin_amdgcn_wmma_f32_16x16x32_bf16(false, a, false, b,
                                                 (short)0, c, false, false);
}

// ---------------------------------------------------------------------------
// fp32 -> bf16 bulk convert (8 elements / thread)
// ---------------------------------------------------------------------------
__global__ void k_f2b(__bf16* __restrict__ dst, const float* __restrict__ src,
                      int n8) {
  int t = blockIdx.x * blockDim.x + threadIdx.x;
  if (t >= n8) return;
  float4 a = ((const float4*)src)[2 * t];
  float4 b = ((const float4*)src)[2 * t + 1];
  v8bf o = {f2bf(a.x), f2bf(a.y), f2bf(a.z), f2bf(a.w),
            f2bf(b.x), f2bf(b.y), f2bf(b.z), f2bf(b.w)};
  ((v8bf*)dst)[t] = o;
}

// ---------------------------------------------------------------------------
// K1: emb_bf[i] = bf16(embedding[x[i]])
// ---------------------------------------------------------------------------
__global__ void k_gather(__bf16* __restrict__ emb, const float* __restrict__ table,
                         const int* __restrict__ x, int N) {
  int t = blockIdx.x * blockDim.x + threadIdx.x;
  if (t >= N * (C / 4)) return;
  int node = t / (C / 4);
  int c4 = (t - node * (C / 4)) << 2;
  float4 v = *(const float4*)(table + (size_t)x[node] * C + c4);
  v4bf o = {f2bf(v.x), f2bf(v.y), f2bf(v.z), f2bf(v.w)};
  *(v4bf*)(emb + (size_t)node * C + c4) = o;
}

// ---------------------------------------------------------------------------
// K2: m[dst] += emb[src]  (fp32 atomics; m is 105 MB, L2-resident)
// ---------------------------------------------------------------------------
__global__ void k_scatter(float* __restrict__ m, const __bf16* __restrict__ emb,
                          const int* __restrict__ src, const int* __restrict__ dst,
                          int E) {
  int t = blockIdx.x * blockDim.x + threadIdx.x;
  int per = C / 4;
  if (t >= E * per) return;
  int e = t / per;
  int c4 = (t - e * per) << 2;
  v4bf v = *(const v4bf*)(emb + (size_t)src[e] * C + c4);
  float* o = m + (size_t)dst[e] * C + c4;
  atomicAdd(o + 0, bf2f(v[0]));
  atomicAdd(o + 1, bf2f(v[1]));
  atomicAdd(o + 2, bf2f(v[2]));
  atomicAdd(o + 3, bf2f(v[3]));
}

// ---------------------------------------------------------------------------
// K3: fused GRU cell.  Per wave: one 16-row stripe, full K in registers.
// Gate algebra: r and z gates accumulate the m-part and emb-part into a
// single accumulator each ( [m|emb] @ [w_ih|w_hh]^T is a K-concat ), so only
// 4 live v8f accumulators -> no spills with launch_bounds(256,1).
// ---------------------------------------------------------------------------
__global__ void __launch_bounds__(256, 1)
k_gru(__bf16* __restrict__ h, const __bf16* __restrict__ m,
      const __bf16* __restrict__ emb,
      const __bf16* __restrict__ w_ih, const __bf16* __restrict__ w_hh, int N) {
  int lane = threadIdx.x & 31;
  int rt = blockIdx.x * WPB + (threadIdx.x >> 5);
  if (rt * 16 >= N) return;
  int r0 = rt * 16;
  int n = lane & 15, mb = (lane >> 4) << 3;

  v16bf am[4], ae[4];
#pragma unroll
  for (int kb = 0; kb < 4; ++kb) {
    am[kb] = load_a(m, C, r0, kb * 32, lane);
    ae[kb] = load_a(emb, C, r0, kb * 32, lane);
  }
#pragma unroll 1
  for (int ct = 0; ct < C / 16; ++ct) {
    int n0 = ct * 16;
    v8f ar = zero8(), az = zero8(), ain = zero8(), ahn = zero8();
#pragma unroll
    for (int kb = 0; kb < 4; ++kb) {
      int k0 = kb * 32;
      ar  = wmma_bf16(am[kb], load_b(w_ih, C, n0,         k0, lane), ar);
      ar  = wmma_bf16(ae[kb], load_b(w_hh, C, n0,         k0, lane), ar);
      az  = wmma_bf16(am[kb], load_b(w_ih, C, n0 + C,     k0, lane), az);
      az  = wmma_bf16(ae[kb], load_b(w_hh, C, n0 + C,     k0, lane), az);
      ain = wmma_bf16(am[kb], load_b(w_ih, C, n0 + 2 * C, k0, lane), ain);
      ahn = wmma_bf16(ae[kb], load_b(w_hh, C, n0 + 2 * C, k0, lane), ahn);
    }
#pragma unroll
    for (int i = 0; i < 8; ++i) {
      size_t idx = (size_t)(r0 + mb + i) * C + n0 + n;
      float e  = bf2f(emb[idx]);
      float r  = 1.f / (1.f + __expf(-ar[i]));
      float z  = 1.f / (1.f + __expf(-az[i]));
      float nc = tanhf(ain[i] + r * ahn[i]);
      h[idx] = f2bf((1.f - z) * nc + z * e);
    }
  }
}

// ---------------------------------------------------------------------------
// K4: wl[g] = h[last node of graph g]   (pure bf16 copy)
// ---------------------------------------------------------------------------
__global__ void k_wl(__bf16* __restrict__ wl, const __bf16* __restrict__ h,
                     int B, int npg) {
  int t = blockIdx.x * blockDim.x + threadIdx.x;
  if (t >= B * (C / 4)) return;
  int g = t / (C / 4);
  int c4 = (t - g * (C / 4)) << 2;
  *(v4bf*)(wl + (size_t)g * C + c4) =
      *(const v4bf*)(h + ((size_t)(g + 1) * npg - 1) * C + c4);
}

// ---------------------------------------------------------------------------
// K5: q1 = w_l @ w1^T   (B x 128, fp32 out for later bias path)
// ---------------------------------------------------------------------------
__global__ void __launch_bounds__(256, 1)
k_q1(float* __restrict__ q1, const __bf16* __restrict__ wl,
     const __bf16* __restrict__ w1, int B) {
  int lane = threadIdx.x & 31;
  int rt = blockIdx.x * WPB + (threadIdx.x >> 5);
  if (rt * 16 >= B) return;
  int r0 = rt * 16;
  int n = lane & 15, mb = (lane >> 4) << 3;
  v16bf a[4];
#pragma unroll
  for (int kb = 0; kb < 4; ++kb) a[kb] = load_a(wl, C, r0, kb * 32, lane);
#pragma unroll 1
  for (int ct = 0; ct < C / 16; ++ct) {
    v8f acc = zero8();
#pragma unroll
    for (int kb = 0; kb < 4; ++kb)
      acc = wmma_bf16(a[kb], load_b(w1, C, ct * 16, kb * 32, lane), acc);
#pragma unroll
    for (int i = 0; i < 8; ++i)
      q1[(size_t)(r0 + mb + i) * C + ct * 16 + n] = acc[i];
  }
}

// ---------------------------------------------------------------------------
// K6: fused attention:  s = sigmoid(q1[batch] + h@w2^T + b2)  (bf16 in LDS)
//     alpha = s@wq^T + bq ;  wg[batch] += alpha * h   (fp32 atomics)
// ---------------------------------------------------------------------------
__global__ void __launch_bounds__(256, 1)
k_attn(float* __restrict__ wg, const __bf16* __restrict__ h,
       const float* __restrict__ q1, const __bf16* __restrict__ w2,
       const float* __restrict__ b2, const __bf16* __restrict__ wq,
       const float* __restrict__ bq, const int* __restrict__ batch, int N) {
  __shared__ __align__(64) __bf16 s[WPB][16 * C];  // 4 KB/wave, 32 KB/block
  int lane = threadIdx.x & 31, wid = threadIdx.x >> 5;
  int rt = blockIdx.x * WPB + wid;
  bool active = (rt * 16 < N);
  int r0 = rt * 16;
  int n = lane & 15, mb = (lane >> 4) << 3;

  if (active) {
    v16bf ah[4];
#pragma unroll
    for (int kb = 0; kb < 4; ++kb) ah[kb] = load_a(h, C, r0, kb * 32, lane);
#pragma unroll 1
    for (int ct = 0; ct < C / 16; ++ct) {
      v8f acc = zero8();
#pragma unroll
      for (int kb = 0; kb < 4; ++kb)
        acc = wmma_bf16(ah[kb], load_b(w2, C, ct * 16, kb * 32, lane), acc);
#pragma unroll
      for (int i = 0; i < 8; ++i) {
        int row = r0 + mb + i;
        float v = acc[i] + b2[ct * 16 + n] +
                  q1[(size_t)batch[row] * C + ct * 16 + n];
        s[wid][(mb + i) * C + ct * 16 + n] = f2bf(1.f / (1.f + __expf(-v)));
      }
    }
  }
  __syncthreads();
  if (active) {
    v16bf as[4];
#pragma unroll
    for (int kb = 0; kb < 4; ++kb) as[kb] = load_a(&s[wid][0], C, 0, kb * 32, lane);
#pragma unroll 1
    for (int ct = 0; ct < C / 16; ++ct) {
      v8f acc = zero8();
#pragma unroll
      for (int kb = 0; kb < 4; ++kb)
        acc = wmma_bf16(as[kb], load_b(wq, C, ct * 16, kb * 32, lane), acc);
#pragma unroll
      for (int i = 0; i < 8; ++i) {
        int row = r0 + mb + i;
        float alpha = acc[i] + bq[ct * 16 + n];
        float hv = bf2f(h[(size_t)row * C + ct * 16 + n]);
        atomicAdd(&wg[(size_t)batch[row] * C + ct * 16 + n], alpha * hv);
      }
    }
  }
}

// ---------------------------------------------------------------------------
// K7: fused head: w = [w_l | w_g] @ wt^T  (K=256) staged bf16 in LDS, then
//     logits = w @ embedding^T  (16 x 2048 per wave, table_bf is L2-resident)
// ---------------------------------------------------------------------------
__global__ void __launch_bounds__(256, 1)
k_final(float* __restrict__ out, const __bf16* __restrict__ wl,
        const __bf16* __restrict__ wg, const __bf16* __restrict__ wt,
        const __bf16* __restrict__ table, int B) {
  __shared__ __align__(64) __bf16 s[WPB][16 * C];
  int lane = threadIdx.x & 31, wid = threadIdx.x >> 5;
  int rt = blockIdx.x * WPB + wid;
  bool active = (rt * 16 < B);
  int r0 = rt * 16;
  int n = lane & 15, mb = (lane >> 4) << 3;

  if (active) {
    v16bf al[4], ag[4];
#pragma unroll
    for (int kb = 0; kb < 4; ++kb) {
      al[kb] = load_a(wl, C, r0, kb * 32, lane);
      ag[kb] = load_a(wg, C, r0, kb * 32, lane);
    }
#pragma unroll 1
    for (int ct = 0; ct < C / 16; ++ct) {
      v8f acc = zero8();
#pragma unroll
      for (int kb = 0; kb < 4; ++kb)  // k in [0,128): w_l half of concat
        acc = wmma_bf16(al[kb], load_b(wt, 2 * C, ct * 16, kb * 32, lane), acc);
#pragma unroll
      for (int kb = 0; kb < 4; ++kb)  // k in [128,256): w_g half of concat
        acc = wmma_bf16(ag[kb], load_b(wt, 2 * C, ct * 16, C + kb * 32, lane), acc);
#pragma unroll
      for (int i = 0; i < 8; ++i)
        s[wid][(mb + i) * C + ct * 16 + n] = f2bf(acc[i]);
    }
  }
  __syncthreads();
  if (active) {
    v16bf aw[4];
#pragma unroll
    for (int kb = 0; kb < 4; ++kb) aw[kb] = load_a(&s[wid][0], C, 0, kb * 32, lane);
#pragma unroll 1
    for (int tt = 0; tt < T / 16; ++tt) {
      v8f acc = zero8();
#pragma unroll
      for (int kb = 0; kb < 4; ++kb)
        acc = wmma_bf16(aw[kb], load_b(table, C, tt * 16, kb * 32, lane), acc);
#pragma unroll
      for (int i = 0; i < 8; ++i)
        out[(size_t)(r0 + mb + i) * T + tt * 16 + n] = acc[i];
    }
  }
}

// ---------------------------------------------------------------------------
// Host launcher (graph-capture safe: only async ops on `stream`)
// ---------------------------------------------------------------------------
extern "C" void kernel_launch(void* const* d_in, const int* in_sizes, int n_in,
                              void* d_out, int out_size, void* d_ws, size_t ws_size,
                              hipStream_t stream) {
  const int*   x     = (const int*)d_in[0];
  const int*   edge  = (const int*)d_in[1];   // [2,E] flat: src then dst
  const int*   batch = (const int*)d_in[2];
  // d_in[3] = num_graphs (device scalar); B derived from out_size instead
  const float* table = (const float*)d_in[4];
  const float* w_ih  = (const float*)d_in[5];
  const float* w_hh  = (const float*)d_in[6];
  const float* w1    = (const float*)d_in[7];
  const float* w2    = (const float*)d_in[8];
  const float* b2    = (const float*)d_in[9];
  const float* wq    = (const float*)d_in[10];
  const float* bq    = (const float*)d_in[11];
  const float* wt    = (const float*)d_in[12];
  float* out = (float*)d_out;

  const int N   = in_sizes[0];
  const int E   = in_sizes[1] / 2;
  const int B   = out_size / T;
  const int NPG = N / B;

  // ---- workspace carve-up (all sizes are multiples of 256 bytes) ----
  char* p = (char*)d_ws;
  auto alloc_f32 = [&](size_t n) { float* r = (float*)p;  p += n * 4; return r; };
  auto alloc_bf  = [&](size_t n) { __bf16* r = (__bf16*)p; p += n * 2; return r; };

  float*  m     = alloc_f32((size_t)N * C);
  float*  q1    = alloc_f32((size_t)B * C);
  float*  wg    = alloc_f32((size_t)B * C);
  __bf16* embb  = alloc_bf((size_t)N * C);
  __bf16* mb_   = alloc_bf((size_t)N * C);
  __bf16* hb    = alloc_bf((size_t)N * C);
  __bf16* wlb   = alloc_bf((size_t)B * C);
  __bf16* wgb   = alloc_bf((size_t)B * C);
  __bf16* wihb  = alloc_bf((size_t)3 * C * C);
  __bf16* whhb  = alloc_bf((size_t)3 * C * C);
  __bf16* w1b   = alloc_bf((size_t)C * C);
  __bf16* w2b   = alloc_bf((size_t)C * C);
  __bf16* wqb   = alloc_bf((size_t)C * C);
  __bf16* wtb   = alloc_bf((size_t)2 * C * C);
  __bf16* tabb  = alloc_bf((size_t)T * C);

  hipMemsetAsync(m,  0, (size_t)N * C * sizeof(float), stream);
  hipMemsetAsync(wg, 0, (size_t)B * C * sizeof(float), stream);

  auto cvt = [&](__bf16* d, const float* s, int n) {
    int n8 = n / 8;
    k_f2b<<<(n8 + 255) / 256, 256, 0, stream>>>(d, s, n8);
  };
  cvt(wihb, w_ih, 3 * C * C);
  cvt(whhb, w_hh, 3 * C * C);
  cvt(w1b,  w1,   C * C);
  cvt(w2b,  w2,   C * C);
  cvt(wqb,  wq,   C * C);
  cvt(wtb,  wt,   2 * C * C);
  cvt(tabb, table, T * C);

  { int tot = N * (C / 4);
    k_gather<<<(tot + 255) / 256, 256, 0, stream>>>(embb, table, x, N); }
  { int tot = E * (C / 4);
    k_scatter<<<(tot + 255) / 256, 256, 0, stream>>>(m, embb, edge, edge + E, E); }
  cvt(mb_, m, N * C);
  { int nt = (N + 15) / 16;
    k_gru<<<(nt + WPB - 1) / WPB, WPB * 32, 0, stream>>>(hb, mb_, embb, wihb, whhb, N); }
  { int tot = B * (C / 4);
    k_wl<<<(tot + 255) / 256, 256, 0, stream>>>(wlb, hb, B, NPG); }
  { int nt = (B + 15) / 16;
    k_q1<<<(nt + WPB - 1) / WPB, WPB * 32, 0, stream>>>(q1, wlb, w1b, B); }
  { int nt = (N + 15) / 16;
    k_attn<<<(nt + WPB - 1) / WPB, WPB * 32, 0, stream>>>(wg, hb, q1, w2b, b2,
                                                          wqb, bq, batch, N); }
  cvt(wgb, wg, B * C);
  { int nt = (B + 15) / 16;
    k_final<<<(nt + WPB - 1) / WPB, WPB * 32, 0, stream>>>(out, wlb, wgb, wtb,
                                                           tabb, B); }
}
